// MultiHeadAttention_57698590655121
// MI455X (gfx1250) — matrix operand order repaired and checked by reading, not compiled, not run on hardware
//
#include <hip/hip_runtime.h>

#define BB 4
#define SS 2048
#define DD 1024
#define HH 16
#define HD 64
#define MM (BB * SS) /* 8192 */

typedef __attribute__((ext_vector_type(16))) _Float16 v16h;
typedef __attribute__((ext_vector_type(8)))  _Float16 v8h;
typedef __attribute__((ext_vector_type(4)))  _Float16 v4h;
typedef __attribute__((ext_vector_type(8)))  float    v8f;

static __device__ __forceinline__ v8f wmma_f16(v16h a, v16h b, v8f c) {
  // D = A(16x32 f16) * B(32x16 f16) + C(16x16 f32)
  return __builtin_amdgcn_wmma_f32_16x16x32_f16(false, a, false, b, (short)0, c,
                                                false, false);
}

// raw hardware base-2 exponential (v_exp_f32): no libm range-fixup code.
// Arguments here are always <= 0; v_exp_f32(very negative) -> 0, which is
// exactly what masked lanes need.
static __device__ __forceinline__ float fast_exp2(float x) {
  return __builtin_amdgcn_exp2f(x);
}

static __device__ __forceinline__ v16h cat16(v8h lo, v8h hi) {
  v16h r;
#pragma unroll
  for (int i = 0; i < 8; ++i) { r[i] = lo[i]; r[8 + i] = hi[i]; }
  return r;
}

// A-fragment: lane (lr) = matrix row, rowp = &A[row][kchunk]. ISA 16-bit A layout:
// halves 0..7 = K{hs*8..hs*8+7}, halves 8..15 = K{16+hs*8..16+hs*8+7}
static __device__ __forceinline__ v16h load_a(const _Float16* rowp, int hs) {
  v8h lo = *(const v8h*)(rowp + hs * 8);
  v8h hi = *(const v8h*)(rowp + 16 + hs * 8);
  return cat16(lo, hi);
}
// B-fragment: lane (lr) = matrix column == a row of the row-major second operand;
// halves 0..7 = K{hs*8..hs*8+7}, halves 8..15 = K{16+hs*8..16+hs*8+7} (same as A).
static __device__ __forceinline__ v16h load_b(const _Float16* rowp, int hs) {
  v8h lo = *(const v8h*)(rowp + hs * 8);
  v8h hi = *(const v8h*)(rowp + 16 + hs * 8);
  return cat16(lo, hi);
}

// ---------------------------------------------------------------- converts ---
__global__ void cvt_f16_kernel(const float* __restrict__ in, _Float16* __restrict__ out) {
  size_t i = ((size_t)blockIdx.x * blockDim.x + threadIdx.x) * 4;
  float4 v = *(const float4*)(in + i);
  v4h o;
  o[0] = (_Float16)v.x; o[1] = (_Float16)v.y; o[2] = (_Float16)v.z; o[3] = (_Float16)v.w;
  *(v4h*)(out + i) = o;
}

// out[n][d] = (f16) in[d][n]   (D x D)
__global__ void transpose_cvt_kernel(const float* __restrict__ in, _Float16* __restrict__ out) {
  __shared__ float tile[32][33];
  int x = blockIdx.x * 32 + threadIdx.x;
  int y = blockIdx.y * 32 + threadIdx.y;
#pragma unroll
  for (int j = 0; j < 32; j += 8)
    tile[threadIdx.y + j][threadIdx.x] = in[(size_t)(y + j) * DD + x];
  __syncthreads();
  int ox = blockIdx.y * 32 + threadIdx.x;
  int oy = blockIdx.x * 32 + threadIdx.y;
#pragma unroll
  for (int j = 0; j < 32; j += 8)
    out[(size_t)(oy + j) * DD + ox] = (_Float16)tile[threadIdx.x][threadIdx.y + j];
}

// ------------------------------------------------------------------- GEMM ----
// R[i][j] = sum_k P[i][k] * Q[j][k]  (both row-major along k), stores R^T:
// out[j][i] (row-major, leading dim I).  Optional f32 output + bias over i.
// Wave computes a 32(i) x 32(j) tile with 4 accumulators.
template <bool F32OUT>
__launch_bounds__(256)
__global__ void gemm_tt_kernel(const _Float16* __restrict__ Pm,
                               const _Float16* __restrict__ Qm,
                               void* __restrict__ outv,
                               const float* __restrict__ bias,
                               int I, int J, int K) {
  const int wave = blockIdx.x * 8 + (threadIdx.x >> 5);
  const int lane = threadIdx.x & 31;
  const int lr = lane & 15;
  const int hs = lane >> 4;
  const int tilesI = I >> 5;
  const int ti = wave % tilesI;
  const int tj = wave / tilesI;
  const int i0 = ti * 32, j0 = tj * 32;

  const _Float16* p0 = Pm + (size_t)(i0 + lr) * K;
  const _Float16* p1 = Pm + (size_t)(i0 + 16 + lr) * K;
  const _Float16* q0p = Qm + (size_t)(j0 + lr) * K;
  const _Float16* q1p = Qm + (size_t)(j0 + 16 + lr) * K;

  v8f c00 = {}, c01 = {}, c10 = {}, c11 = {};
  for (int k = 0; k < K; k += 32) {
    v16h a0 = load_a(p0 + k, hs);
    v16h a1 = load_a(p1 + k, hs);
    v16h b0 = load_b(q0p + k, hs);
    v16h b1 = load_b(q1p + k, hs);
    c00 = wmma_f16(a0, b0, c00);
    c01 = wmma_f16(a0, b1, c01);
    c10 = wmma_f16(a1, b0, c10);
    c11 = wmma_f16(a1, b1, c11);
  }

  auto store_tile = [&](v8f c, int it, int jt) {
    size_t row = (size_t)(j0 + jt * 16 + lr);
    int col = i0 + it * 16 + hs * 8;
    if (F32OUT) {
      float* out = (float*)outv;
      float4 lo, hi;
      lo.x = c[0] + bias[col + 0]; lo.y = c[1] + bias[col + 1];
      lo.z = c[2] + bias[col + 2]; lo.w = c[3] + bias[col + 3];
      hi.x = c[4] + bias[col + 4]; hi.y = c[5] + bias[col + 5];
      hi.z = c[6] + bias[col + 6]; hi.w = c[7] + bias[col + 7];
      *(float4*)(out + row * I + col) = lo;
      *(float4*)(out + row * I + col + 4) = hi;
    } else {
      _Float16* out = (_Float16*)outv;
      v8h hv;
#pragma unroll
      for (int r = 0; r < 8; ++r) hv[r] = (_Float16)c[r];
      *(v8h*)(out + row * I + col) = hv;
    }
  };
  store_tile(c00, 0, 0);
  store_tile(c01, 0, 1);
  store_tile(c10, 1, 0);
  store_tile(c11, 1, 1);
}

// -------------------------------------------------------------- attention ----
// One wave per (b, h, 16-row q tile). Flash-style online softmax over 32-wide
// kv blocks, computed transposed (S^T = K * Q^T) so per-q statistics are
// per-lane. Scores are kept in the log2 domain (scale * log2e folded into one
// constant) so the raw hw v_exp_f32 (base-2) is used directly. The causal mask
// is only applied in the <=2 diagonal tail blocks (MASKED=true instantiation);
// the main loop is mask-free.
template <bool MASKED>
static __device__ __forceinline__ void attn_block(
    int kv0, int qg, int hs, int lr,
    const _Float16* Kb, const _Float16* Vb, _Float16* plds,
    v16h bq0, v16h bq1,
    v8f& y0, v8f& y1, v8f& y2, v8f& y3,
    float& m_run, float& l_run) {
  const float c1 = 0.18033688011112042f;  // (1/sqrt(64)) * log2(e)
  const _Float16* kr0 = Kb + (size_t)(kv0 + lr) * DD;
  const _Float16* kr1 = Kb + (size_t)(kv0 + 16 + lr) * DD;
  v8f s0 = {}, s1 = {};
  s0 = wmma_f16(load_a(kr0, hs), bq0, s0);
  s0 = wmma_f16(load_a(kr0 + 32, hs), bq1, s0);
  s1 = wmma_f16(load_a(kr1, hs), bq0, s1);
  s1 = wmma_f16(load_a(kr1 + 32, hs), bq1, s1);

  // lane holds S^T[kv][q] for q = qg, kv rows hs*8+r and 16+hs*8+r
  float sv[16];
#pragma unroll
  for (int r = 0; r < 8; ++r) {
    sv[r] = s0[r] * c1;
    sv[8 + r] = s1[r] * c1;
    if (MASKED) {
      if (kv0 + hs * 8 + r > qg) sv[r] = -1e30f;
      if (kv0 + 16 + hs * 8 + r > qg) sv[8 + r] = -1e30f;
    }
  }
  float mloc = sv[0];
#pragma unroll
  for (int i = 1; i < 16; ++i) mloc = fmaxf(mloc, sv[i]);
  mloc = fmaxf(mloc, __shfl_xor(mloc, 16, 32));  // combine lane pair (q split)
  float mnew = fmaxf(m_run, mloc);
  float psum = 0.0f;
#pragma unroll
  for (int i = 0; i < 16; ++i) {
    sv[i] = fast_exp2(sv[i] - mnew);
    psum += sv[i];
  }
  psum += __shfl_xor(psum, 16, 32);
  float alpha = fast_exp2(m_run - mnew);
  l_run = l_run * alpha + psum;
  m_run = mnew;

  // pack P to LDS as [q][kv] (f16), padded stride 40 halves
  v8h pa, pb;
#pragma unroll
  for (int r = 0; r < 8; ++r) { pa[r] = (_Float16)sv[r]; pb[r] = (_Float16)sv[8 + r]; }
  *(v8h*)(plds + lr * 40 + hs * 8) = pa;
  *(v8h*)(plds + lr * 40 + 16 + hs * 8) = pb;
  __syncthreads();

  // per-row rescale of accumulators (row r of C-layout is q = hs*8 + r)
  float al[8];
#pragma unroll
  for (int r = 0; r < 8; ++r) al[r] = __shfl(alpha, hs * 8 + r, 32);
#pragma unroll
  for (int r = 0; r < 8; ++r) {
    y0[r] *= al[r]; y1[r] *= al[r]; y2[r] *= al[r]; y3[r] *= al[r];
  }

  // P (16q x 32kv) as A-fragment from LDS; V^T rows as B-fragments
  v16h ap = load_a(plds + lr * 40, hs);
  y0 = wmma_f16(ap, load_b(Vb + (size_t)(0 * 16 + lr) * MM + kv0, hs), y0);
  y1 = wmma_f16(ap, load_b(Vb + (size_t)(1 * 16 + lr) * MM + kv0, hs), y1);
  y2 = wmma_f16(ap, load_b(Vb + (size_t)(2 * 16 + lr) * MM + kv0, hs), y2);
  y3 = wmma_f16(ap, load_b(Vb + (size_t)(3 * 16 + lr) * MM + kv0, hs), y3);
  __syncthreads();
}

__launch_bounds__(32)
__global__ void attn_kernel(const _Float16* __restrict__ Qh,
                            const _Float16* __restrict__ Kh,
                            const _Float16* __restrict__ Vt,
                            _Float16* __restrict__ Oh) {
  __shared__ __align__(16) _Float16 plds[16 * 40];  // [q][kv], stride 40 halves
  const int lane = threadIdx.x & 31;
  const int lr = lane & 15;
  const int hs = lane >> 4;
  const int q0 = blockIdx.x * 16;
  const int h = blockIdx.y;
  const int b = blockIdx.z;

  const _Float16* Qb = Qh + (size_t)b * SS * DD + (size_t)h * HD;
  const _Float16* Kb = Kh + (size_t)b * SS * DD + (size_t)h * HD;
  const _Float16* Vb = Vt + (size_t)h * HD * MM + (size_t)b * SS;

  // Q^T B-fragments (K-dim = head dim, two chunks of 32), loaded once.
  const _Float16* qrow = Qb + (size_t)(q0 + lr) * DD;
  v16h bq0 = load_b(qrow, hs);
  v16h bq1 = load_b(qrow + 32, hs);

  v8f y0 = {}, y1 = {}, y2 = {}, y3 = {};  // Y[16q x 64hd], C-layout
  float m_run = -1e30f, l_run = 0.0f;
  const int qg = q0 + lr;  // this lane's q column

  const int nfull = q0 >> 5;             // blocks with kv0+31 <= q0: mask-free
  const int nblk = (q0 + 16 + 31) >> 5;  // total blocks (<=2 masked tail)
  int blk = 0;
  for (; blk < nfull; ++blk)
    attn_block<false>(blk * 32, qg, hs, lr, Kb, Vb, plds, bq0, bq1,
                      y0, y1, y2, y3, m_run, l_run);
  for (; blk < nblk; ++blk)
    attn_block<true>(blk * 32, qg, hs, lr, Kb, Vb, plds, bq0, bq1,
                     y0, y1, y2, y3, m_run, l_run);

  float linv = 1.0f / l_run;
  float li[8];
#pragma unroll
  for (int r = 0; r < 8; ++r) li[r] = __shfl(linv, hs * 8 + r, 32);

  _Float16* Ob = Oh + (size_t)b * SS * DD + (size_t)h * HD;
#pragma unroll
  for (int r = 0; r < 8; ++r) {
    size_t row = (size_t)(q0 + hs * 8 + r) * DD;
    Ob[row + 0 * 16 + lr] = (_Float16)(y0[r] * li[r]);
    Ob[row + 1 * 16 + lr] = (_Float16)(y1[r] * li[r]);
    Ob[row + 2 * 16 + lr] = (_Float16)(y2[r] * li[r]);
    Ob[row + 3 * 16 + lr] = (_Float16)(y3[r] * li[r]);
  }
}

// ------------------------------------------------------------------ launch ---
extern "C" void kernel_launch(void* const* d_in, const int* in_sizes, int n_in,
                              void* d_out, int out_size, void* d_ws, size_t ws_size,
                              hipStream_t stream) {
  (void)in_sizes; (void)n_in; (void)out_size; (void)ws_size;
  const float* x  = (const float*)d_in[0];
  const float* Wq = (const float*)d_in[1];
  const float* Wk = (const float*)d_in[2];
  const float* Wv = (const float*)d_in[3];
  const float* Wo = (const float*)d_in[4];
  const float* bo = (const float*)d_in[5];
  float* out = (float*)d_out;

  char* ws = (char*)d_ws;
  const size_t XH = (size_t)MM * DD * 2;  // 16 MB
  const size_t WT = (size_t)DD * DD * 2;  //  2 MB
  _Float16* Xh  = (_Float16*)(ws);
  _Float16* WqT = (_Float16*)(ws + XH);
  _Float16* WkT = (_Float16*)(ws + XH + 1 * WT);
  _Float16* WvT = (_Float16*)(ws + XH + 2 * WT);
  _Float16* WoT = (_Float16*)(ws + XH + 3 * WT);
  _Float16* Qh  = (_Float16*)(ws + XH + 4 * WT);
  _Float16* Kh  = (_Float16*)(ws + 2 * XH + 4 * WT);
  _Float16* Vt  = (_Float16*)(ws + 3 * XH + 4 * WT);
  _Float16* Oh  = Xh;  // Xh is dead after the V projection; reuse for attn out

  cvt_f16_kernel<<<(MM * DD / 4) / 256, 256, 0, stream>>>(x, Xh);
  dim3 tb(32, 8), tg(DD / 32, DD / 32);
  transpose_cvt_kernel<<<tg, tb, 0, stream>>>(Wq, WqT);
  transpose_cvt_kernel<<<tg, tb, 0, stream>>>(Wk, WkT);
  transpose_cvt_kernel<<<tg, tb, 0, stream>>>(Wv, WvT);
  transpose_cvt_kernel<<<tg, tb, 0, stream>>>(Wo, WoT);

  // Q[m][n], K[m][n] (f16): R = W^T x^T stored transposed -> contiguous stores
  gemm_tt_kernel<false><<<1024, 256, 0, stream>>>(WqT, Xh, (void*)Qh, nullptr, DD, MM, DD);
  gemm_tt_kernel<false><<<1024, 256, 0, stream>>>(WkT, Xh, (void*)Kh, nullptr, DD, MM, DD);
  // Vt[n][m] (f16): swap operands so the transposed store lands as V^T
  gemm_tt_kernel<false><<<1024, 256, 0, stream>>>(Xh, WvT, (void*)Vt, nullptr, MM, DD, DD);

  attn_kernel<<<dim3(SS / 16, HH, BB), 32, 0, stream>>>(Qh, Kh, Vt, Oh);

  // out[m][n] (f32) = O @ Wo + bo
  gemm_tt_kernel<true><<<1024, 256, 0, stream>>>(WoT, Oh, (void*)out, bo, DD, MM, DD);
}
